// MotionTransformer_67731634258770
// MI455X (gfx1250) — compile-verified, matrix operand order checked by reading, not run on hardware
//
#include <hip/hip_runtime.h>

// MI455X (gfx1250) implementation: bf16 WMMA GEMMs (v_wmma_f32_16x16x32_bf16)
// for all matmuls; the model is ~1 TFLOP at ~3000 FLOP/byte -> matrix-core
// bound, so fp32 inputs are converted to bf16 and accumulated in f32.
// Staging uses guard-free vectorized fast paths for interior tiles so the
// per-K-tile instruction budget is dominated by the 8 WMMAs, not exec masking.

typedef __attribute__((ext_vector_type(16))) __bf16 v16bf;
typedef __attribute__((ext_vector_type(8)))  float  v8f;

namespace {

constexpr int kB   = 128;   // batch
constexpr int kT   = 196;   // tokens
constexpr int kD   = 512;   // model dim
constexpr int kTE  = 2048;  // time-embed dim
constexpr int kFF  = 1024;  // ffn dim
constexpr int kLT  = 256;   // text latent dim
constexpr int kL   = 8;     // layers
constexpr int kH   = 8;     // heads
constexpr int kDh  = 64;    // head dim
constexpr int kN   = 77;    // text tokens
constexpr int kIF  = 263;   // in/out features
constexpr int kBT  = kB * kT;

__device__ __forceinline__ unsigned short f2bf(float x) {
  unsigned u = __float_as_uint(x);
  u += 0x7FFFu + ((u >> 16) & 1u);          // round-to-nearest-even
  return (unsigned short)(u >> 16);
}
__device__ __forceinline__ float siluf(float x) { return x / (1.f + __expf(-x)); }
__device__ __forceinline__ float geluf(float x) { return 0.5f * x * (1.f + erff(x * 0.70710678118f)); }

union Frag { uint4 q[2]; v16bf v; };

// ---------------------------------------------------------------------------
// Generic strided/batched GEMM: C = epilogue(A[M,K] @ B[K,N]).
// A element: A[offA + m*sAm + k*sAk]; B element: B[offB + k*sBk + n*sBn].
// Batch z: off* = (z/bDiv)*b*1 + (z%bDiv)*b*2 (two-level batch for (b,h)).
// Epilogue: v = (dot + bias[n]) * rowScale[m]; act; v += add[cIdx]; store.
// ---------------------------------------------------------------------------
template <int BM, int BN, int WM, int WN>
__global__ __launch_bounds__((BM / WM) * (BN / WN) * 32)
void k_gemm(const float* __restrict__ A, const float* __restrict__ Bm,
            float* __restrict__ C, int M, int N, int K,
            long sAm, long sAk, long sBk, long sBn, long sCm,
            int bDiv, long bA1, long bA2, long bB1, long bB2, long bC1, long bC2,
            const float* __restrict__ bias, const float* __restrict__ add,
            const float* __restrict__ rowScale, int act) {
  constexpr int BK = 32, LDT = BK + 8;            // +8 bf16 pad: conflict-free 16B rows
  constexpr int NT = (BM / WM) * (BN / WN) * 32;
  constexpr int TM = WM / 16, TN = WN / 16;
  constexpr int WAVES_N = BN / WN;
  __shared__ alignas(16) unsigned short sA[BM * LDT];
  __shared__ alignas(16) unsigned short sB[BN * LDT];

  const int tid = threadIdx.x, lane = tid & 31, wave = tid >> 5;
  const int wm = wave / WAVES_N, wn = wave % WAVES_N;
  const int z = blockIdx.z;
  const long offA = (long)(z / bDiv) * bA1 + (long)(z % bDiv) * bA2;
  const long offB = (long)(z / bDiv) * bB1 + (long)(z % bDiv) * bB2;
  const long offC = (long)(z / bDiv) * bC1 + (long)(z % bDiv) * bC2;
  const int m0 = blockIdx.y * BM, n0 = blockIdx.x * BN;

  // Uniform (SGPR) staging-mode selection: vectorize along contiguous dim.
  const bool aVecK = (sAk == 1) && ((sAm & 3) == 0) && ((offA & 3) == 0);
  const bool aVecM = (sAm == 1) && ((sAk & 3) == 0) && ((offA & 3) == 0);
  const bool bVecN = (sBn == 1) && ((sBk & 3) == 0) && ((offB & 3) == 0);
  const bool mFull = (m0 + BM <= M);
  const bool nFull = (n0 + BN <= N);

  v8f acc[TM][TN] = {};

  for (int k0 = 0; k0 < K; k0 += BK) {
    const bool kFull = (k0 + BK <= K);
    // ---------------- stage A tile: sA[m][k], bf16 ----------------
    if (kFull && mFull && aVecK) {
      // global_load_b128 + ds_store_b64, no guards
      for (int t = tid; t < BM * (BK / 4); t += NT) {
        int mm = t >> 3, kq = t & 7;                  // BK/4 == 8
        const float4 v = *(const float4*)(A + offA + (long)(m0 + mm) * sAm + (k0 + kq * 4));
        ushort4 h; h.x = f2bf(v.x); h.y = f2bf(v.y); h.z = f2bf(v.z); h.w = f2bf(v.w);
        *(ushort4*)&sA[mm * LDT + kq * 4] = h;
      }
    } else if (kFull && mFull && aVecM) {
      // contiguous along M (k^T @ v einsum): float4 over 4 rows
      for (int t = tid; t < (BM / 4) * BK; t += NT) {
        int kk = t / (BM / 4), mq = t % (BM / 4);
        const float4 v = *(const float4*)(A + offA + (long)(m0 + mq * 4) + (long)(k0 + kk) * sAk);
        sA[(mq * 4 + 0) * LDT + kk] = f2bf(v.x);
        sA[(mq * 4 + 1) * LDT + kk] = f2bf(v.y);
        sA[(mq * 4 + 2) * LDT + kk] = f2bf(v.z);
        sA[(mq * 4 + 3) * LDT + kk] = f2bf(v.w);
      }
    } else {
      for (int t = tid; t < BM * BK; t += NT) {
        int mm = t / BK, kk = t % BK;
        int gm = m0 + mm, gk = k0 + kk;
        float v = (gm < M && gk < K) ? A[offA + (long)gm * sAm + (long)gk * sAk] : 0.f;
        sA[mm * LDT + kk] = f2bf(v);
      }
    }
    // ---------------- stage B tile transposed: sB[n][k], bf16 ----------------
    if (kFull && nFull && bVecN) {
      for (int t = tid; t < (BN / 4) * BK; t += NT) {
        int kk = t / (BN / 4), nq = t % (BN / 4);
        const float4 v = *(const float4*)(Bm + offB + (long)(k0 + kk) * sBk + (n0 + nq * 4));
        sB[(nq * 4 + 0) * LDT + kk] = f2bf(v.x);
        sB[(nq * 4 + 1) * LDT + kk] = f2bf(v.y);
        sB[(nq * 4 + 2) * LDT + kk] = f2bf(v.z);
        sB[(nq * 4 + 3) * LDT + kk] = f2bf(v.w);
      }
    } else {
      for (int t = tid; t < BN * BK; t += NT) {
        int kk = t / BN, nn = t % BN;
        int gn = n0 + nn, gk = k0 + kk;
        float v = (gn < N && gk < K) ? Bm[offB + (long)gk * sBk + (long)gn * sBn] : 0.f;
        sB[nn * LDT + kk] = f2bf(v);
      }
    }
    if (k0 + BK < K) {  // speculative prefetch of next K tile -> global_prefetch_b8
      if (mFull) __builtin_prefetch(A + offA + (long)(m0 + (tid & (BM - 1))) * sAm + (long)(k0 + BK) * sAk, 0, 1);
      if (nFull) __builtin_prefetch(Bm + offB + (long)(k0 + BK) * sBk + (long)(n0 + (tid & (BN - 1))) * sBn, 0, 1);
    }
    __syncthreads();

    // ISA 16-bit A layout: lane&15 = row, lanes>=16 take K-base 8;
    // element e covers K in two 8-wide runs {kb..kb+7} and {kb+16..kb+23}.
    const int kb = (lane >> 4) * 8;
    Frag fa[TM], fb[TN];
#pragma unroll
    for (int i = 0; i < TM; ++i) {
      const unsigned short* p = &sA[(wm * WM + i * 16 + (lane & 15)) * LDT + kb];
      fa[i].q[0] = *(const uint4*)p;
      fa[i].q[1] = *(const uint4*)(p + 16);
    }
#pragma unroll
    for (int j = 0; j < TN; ++j) {
      const unsigned short* p = &sB[(wn * WN + j * 16 + (lane & 15)) * LDT + kb];
      fb[j].q[0] = *(const uint4*)p;
      fb[j].q[1] = *(const uint4*)(p + 16);
    }
#pragma unroll
    for (int i = 0; i < TM; ++i)
#pragma unroll
      for (int j = 0; j < TN; ++j)
        acc[i][j] = __builtin_amdgcn_wmma_f32_16x16x32_bf16(
            false, fa[i].v, false, fb[j].v, (short)0, acc[i][j], false, false);
    __syncthreads();
  }

  // f32 C layout: VGPR r -> M=r (lanes 0-15) / M=r+8 (lanes 16-31); N=lane&15.
  const int rofs = (lane >> 4) * 8;
  if (mFull && nFull) {
    // guard-free epilogue (common case)
#pragma unroll
    for (int i = 0; i < TM; ++i)
#pragma unroll
      for (int j = 0; j < TN; ++j) {
        int n = n0 + wn * WN + j * 16 + (lane & 15);
#pragma unroll
        for (int r = 0; r < 8; ++r) {
          int m = m0 + wm * WM + i * 16 + rofs + r;
          float v = acc[i][j][r];
          if (bias) v += bias[n];
          if (rowScale) v *= rowScale[m];
          if (act == 1) v = siluf(v);
          else if (act == 2) v = geluf(v);
          long ci = offC + (long)m * sCm + n;
          if (add) v += add[ci];
          C[ci] = v;
        }
      }
  } else {
#pragma unroll
    for (int i = 0; i < TM; ++i)
#pragma unroll
      for (int j = 0; j < TN; ++j) {
        int n = n0 + wn * WN + j * 16 + (lane & 15);
#pragma unroll
        for (int r = 0; r < 8; ++r) {
          int m = m0 + wm * WM + i * 16 + rofs + r;
          if (m < M && n < N) {
            float v = acc[i][j][r];
            if (bias) v += bias[n];
            if (rowScale) v *= rowScale[m];
            if (act == 1) v = siluf(v);
            else if (act == 2) v = geluf(v);
            long ci = offC + (long)m * sCm + n;
            if (add) v += add[ci];
            C[ci] = v;
          }
        }
      }
  }
}

// ---------------------------------------------------------------------------
// Elementwise / reduction helper kernels (wave32-aware).
// ---------------------------------------------------------------------------
__global__ void k_timestep(const int* __restrict__ t, float* __restrict__ te, int B) {
  int i = blockIdx.x * blockDim.x + threadIdx.x;
  if (i >= B * 256) return;
  int b = i / 256, j = i % 256;
  float freq = __expf(-9.210340371976184f * (float)j / 256.f);  // ln(10000)
  float a = (float)t[b] * freq;
  te[(size_t)b * 512 + j] = cosf(a);
  te[(size_t)b * 512 + 256 + j] = sinf(a);
}

__global__ void k_silu(const float* __restrict__ x, float* __restrict__ o, long n) {
  long i = (long)blockIdx.x * blockDim.x + threadIdx.x;
  if (i < n) o[i] = siluf(x[i]);
}

__global__ void k_add_seq(float* __restrict__ h, const float* __restrict__ seq,
                          long n, int T, int D) {
  long i = (long)blockIdx.x * blockDim.x + threadIdx.x;
  if (i >= n) return;
  int d = (int)(i % D);
  int t = (int)((i / D) % T);
  h[i] += seq[(size_t)t * D + d];
}

__global__ void k_layernorm(const float* __restrict__ x, float* __restrict__ o,
                            const float* __restrict__ w, const float* __restrict__ b,
                            int rows, int dim) {
  int row = blockIdx.x * (blockDim.x >> 5) + (threadIdx.x >> 5);
  if (row >= rows) return;
  int lane = threadIdx.x & 31;
  const float* xr = x + (size_t)row * dim;
  float s = 0.f, ss = 0.f;
  for (int c = lane; c < dim; c += 32) { float v = xr[c]; s += v; ss += v * v; }
  for (int m = 16; m; m >>= 1) { s += __shfl_xor(s, m, 32); ss += __shfl_xor(ss, m, 32); }
  float mean = s / dim;
  float rs = rsqrtf(ss / dim - mean * mean + 1e-5f);
  float* orow = o + (size_t)row * dim;
  for (int c = lane; c < dim; c += 32) orow[c] = (xr[c] - mean) * rs * w[c] + b[c];
}

// softmax over each contiguous 64-wide head chunk (q path)
__global__ void k_softmax_head(float* __restrict__ q, long chunks) {
  long chunk = (long)blockIdx.x * (blockDim.x >> 5) + (threadIdx.x >> 5);
  if (chunk >= chunks) return;
  int lane = threadIdx.x & 31;
  float* p = q + chunk * 64;
  float a = p[lane], b = p[lane + 32];
  float mx = fmaxf(a, b);
  for (int m = 16; m; m >>= 1) mx = fmaxf(mx, __shfl_xor(mx, m, 32));
  a = __expf(a - mx); b = __expf(b - mx);
  float s = a + b;
  for (int m = 16; m; m >>= 1) s += __shfl_xor(s, m, 32);
  float inv = 1.f / s;
  p[lane] = a * inv;
  p[lane + 32] = b * inv;
}

// softmax over the token axis (axis=1) per (b, channel), optional mask bias
__global__ void k_softmax_seq(float* __restrict__ k, const float* __restrict__ mask,
                              int B, int Tk, int C) {
  long idx = (long)blockIdx.x * blockDim.x + threadIdx.x;
  if (idx >= (long)B * C) return;
  int b = (int)(idx / C), c = (int)(idx % C);
  float* base = k + (size_t)b * Tk * C + c;
  float mx = -3.4e38f;
  for (int n = 0; n < Tk; ++n) {
    float l = base[(size_t)n * C];
    if (mask) l += (1.f - mask[(size_t)b * Tk + n]) * -1000000.f;
    mx = fmaxf(mx, l);
  }
  float s = 0.f;
  for (int n = 0; n < Tk; ++n) {
    float l = base[(size_t)n * C];
    if (mask) l += (1.f - mask[(size_t)b * Tk + n]) * -1000000.f;
    s += __expf(l - mx);
  }
  float inv = 1.f / s;
  for (int n = 0; n < Tk; ++n) {
    float l = base[(size_t)n * C];
    if (mask) l += (1.f - mask[(size_t)b * Tk + n]) * -1000000.f;
    base[(size_t)n * C] = __expf(l - mx) * inv;
  }
}

// z = silu( LN(y)*(1+scale) + shift ), scale/shift = e[b, 0:D]/e[b, D:2D]
__global__ void k_styl_mod(const float* __restrict__ y, const float* __restrict__ e,
                           const float* __restrict__ nw, const float* __restrict__ nb,
                           float* __restrict__ z, int rows, int T) {
  int row = blockIdx.x * (blockDim.x >> 5) + (threadIdx.x >> 5);
  if (row >= rows) return;
  int lane = threadIdx.x & 31;
  int b = row / T;
  const float* yr = y + (size_t)row * kD;
  float vv[16];
  float s = 0.f, ss = 0.f;
#pragma unroll
  for (int i = 0; i < 16; ++i) {
    float v = yr[lane + 32 * i];
    vv[i] = v; s += v; ss += v * v;
  }
  for (int m = 16; m; m >>= 1) { s += __shfl_xor(s, m, 32); ss += __shfl_xor(ss, m, 32); }
  float mean = s / kD;
  float rs = rsqrtf(ss / kD - mean * mean + 1e-5f);
  const float* eb = e + (size_t)b * 2 * kD;
  float* zr = z + (size_t)row * kD;
#pragma unroll
  for (int i = 0; i < 16; ++i) {
    int c = lane + 32 * i;
    float hn = (vv[i] - mean) * rs * nw[c] + nb[c];
    float v = hn * (1.f + eb[c]) + eb[kD + c];
    zr[c] = siluf(v);
  }
}

}  // namespace

// ---------------------------------------------------------------------------
// Host orchestration
// ---------------------------------------------------------------------------
extern "C" void kernel_launch(void* const* d_in, const int* in_sizes, int n_in,
                              void* d_out, int out_size, void* d_ws, size_t ws_size,
                              hipStream_t stream) {
  (void)in_sizes; (void)n_in; (void)out_size; (void)ws_size;
  const float* x       = (const float*)d_in[0];
  const int*   tsteps  = (const int*)  d_in[1];
  const float* mask    = (const float*)d_in[2];   // (B,T,1) -> flat B*T
  const float* xf_proj = (const float*)d_in[3];
  const float* xf_out  = (const float*)d_in[4];
  // layer params (leading L dim each), in setup_inputs dict order:
  const float* sa_nw = (const float*)d_in[5];  const float* sa_nb = (const float*)d_in[6];
  const float* sa_qw = (const float*)d_in[7];  const float* sa_qb = (const float*)d_in[8];
  const float* sa_kw = (const float*)d_in[9];  const float* sa_kb = (const float*)d_in[10];
  const float* sa_vw = (const float*)d_in[11]; const float* sa_vb = (const float*)d_in[12];
  const float* sa_ew = (const float*)d_in[13]; const float* sa_eb = (const float*)d_in[14];
  const float* sa_onw= (const float*)d_in[15]; const float* sa_onb= (const float*)d_in[16];
  const float* sa_ow = (const float*)d_in[17]; const float* sa_ob = (const float*)d_in[18];
  const float* ca_nw = (const float*)d_in[19]; const float* ca_nb = (const float*)d_in[20];
  const float* ca_tnw= (const float*)d_in[21]; const float* ca_tnb= (const float*)d_in[22];
  const float* ca_qw = (const float*)d_in[23]; const float* ca_qb = (const float*)d_in[24];
  const float* ca_kw = (const float*)d_in[25]; const float* ca_kb = (const float*)d_in[26];
  const float* ca_vw = (const float*)d_in[27]; const float* ca_vb = (const float*)d_in[28];
  const float* ca_ew = (const float*)d_in[29]; const float* ca_eb = (const float*)d_in[30];
  const float* ca_onw= (const float*)d_in[31]; const float* ca_onb= (const float*)d_in[32];
  const float* ca_ow = (const float*)d_in[33]; const float* ca_ob = (const float*)d_in[34];
  const float* f1w   = (const float*)d_in[35]; const float* f1b   = (const float*)d_in[36];
  const float* f2w   = (const float*)d_in[37]; const float* f2b   = (const float*)d_in[38];
  const float* fn_ew = (const float*)d_in[39]; const float* fn_eb = (const float*)d_in[40];
  const float* fn_onw= (const float*)d_in[41]; const float* fn_onb= (const float*)d_in[42];
  const float* fn_ow = (const float*)d_in[43]; const float* fn_ob = (const float*)d_in[44];
  const float* jw  = (const float*)d_in[45]; const float* jb  = (const float*)d_in[46];
  const float* t1w = (const float*)d_in[47]; const float* t1b = (const float*)d_in[48];
  const float* t2w = (const float*)d_in[49]; const float* t2b = (const float*)d_in[50];
  const float* seq = (const float*)d_in[51];
  const float* ow  = (const float*)d_in[52]; const float* ob  = (const float*)d_in[53];
  float* out = (float*)d_out;

  // workspace carve
  float* p = (float*)d_ws;
  auto carve = [&](size_t n) { float* r = p; p += n; return r; };
  float* wte   = carve((size_t)kB * kD);
  float* wth   = carve((size_t)kB * kTE);
  float* wemb  = carve((size_t)kB * kTE);
  float* wsemb = carve((size_t)kB * kTE);
  float* we    = carve((size_t)kB * 2 * kD);
  const size_t HT = (size_t)kBT * kD;
  float* wh  = carve(HT);
  float* wxn = carve(HT);   // also reused as attention output y
  float* wq  = carve(HT);   // also reused as stylization z
  float* wk  = carve(HT);   // wk..wv contiguous -> FFN hidden buffer
  float* wv  = carve(HT);
  float* watt= carve((size_t)kB * kH * kDh * kDh);
  float* wxfn= carve((size_t)kB * kN * kLT);
  float* wff = wk;          // B*T*1024 spans wk+wv
  (void)wv;

  hipStream_t s = stream;
  // plain row-major single GEMM: A[M,K] @ B[K,N] (+bias,*rowScale,act,+add)
  auto gemm = [&](const float* A, const float* B, float* C, int M, int N, int K,
                  const float* bias, const float* add, const float* rs, int act) {
    dim3 g((N + 127) / 128, (M + 127) / 128, 1);
    k_gemm<128, 128, 64, 32><<<g, dim3(256), 0, s>>>(
        A, B, C, M, N, K, (long)K, 1L, (long)N, 1L, (long)N,
        1, 0L, 0L, 0L, 0L, 0L, 0L, bias, add, rs, act);
  };
  // strided batched GEMM (64x64 tiles) for per-(b,h) linear-attention einsums
  auto gemmB = [&](const float* A, const float* B, float* C, int M, int N, int K,
                   long sAm, long sAk, long sBk, long sBn, long sCm,
                   int batch, int bDiv, long bA1, long bA2, long bB1, long bB2,
                   long bC1, long bC2) {
    dim3 g((N + 63) / 64, (M + 63) / 64, batch);
    k_gemm<64, 64, 32, 32><<<g, dim3(128), 0, s>>>(
        A, B, C, M, N, K, sAm, sAk, sBk, sBn, sCm,
        bDiv, bA1, bA2, bB1, bB2, bC1, bC2,
        (const float*)nullptr, (const float*)nullptr, (const float*)nullptr, 0);
  };
  auto ln = [&](const float* xi, float* o, const float* w, const float* b, int rows, int dim) {
    k_layernorm<<<(rows + 7) / 8, 256, 0, s>>>(xi, o, w, b, rows, dim);
  };
  auto styl = [&](const float* y, const float* e, const float* nw2, const float* nb2, float* z) {
    k_styl_mod<<<(kBT + 7) / 8, 256, 0, s>>>(y, e, nw2, nb2, z, kBT, kT);
  };

  // ---- timestep embedding + emb MLP ----
  k_timestep<<<(kB * 256 + 255) / 256, 256, 0, s>>>(tsteps, wte, kB);
  gemm(wte, t1w, wth, kB, kTE, kD, t1b, nullptr, nullptr, 1);            // silu
  gemm(wth, t2w, wemb, kB, kTE, kTE, t2b, xf_proj, nullptr, 0);          // + xf_proj
  k_silu<<<(int)(((long)kB * kTE + 255) / 256), 256, 0, s>>>(wemb, wsemb, (long)kB * kTE);

  // ---- input projection + positional sequence ----
  gemm(x, jw, wh, kBT, kD, kIF, jb, nullptr, nullptr, 0);
  k_add_seq<<<(int)((HT + 255) / 256), 256, 0, s>>>(wh, seq, (long)HT, kT, kD);

  const long DD = (long)kD * kD, E2 = (long)kTE * 2 * kD, LD = (long)kLT * kD;
  for (int l = 0; l < kL; ++l) {
    // ================= linear temporal self-attention =================
    ln(wh, wxn, sa_nw + (size_t)l * kD, sa_nb + (size_t)l * kD, kBT, kD);
    gemm(wxn, sa_qw + l * DD, wq, kBT, kD, kD, sa_qb + (size_t)l * kD, nullptr, nullptr, 0);
    k_softmax_head<<<(int)(((long)kBT * kH + 7) / 8), 256, 0, s>>>(wq, (long)kBT * kH);
    gemm(wxn, sa_kw + l * DD, wk, kBT, kD, kD, sa_kb + (size_t)l * kD, nullptr, nullptr, 0);
    k_softmax_seq<<<((kB * kD) + 255) / 256, 256, 0, s>>>(wk, mask, kB, kT, kD);
    gemm(wxn, sa_vw + l * DD, wv, kBT, kD, kD, sa_vb + (size_t)l * kD, nullptr, mask, 0);
    // att[b,h] = k^T (dh x T) @ v (T x dh)
    gemmB(wk, wv, watt, kDh, kDh, kT, 1L, (long)kD, (long)kD, 1L, (long)kDh,
          kB * kH, kH, (long)kT * kD, (long)kDh, (long)kT * kD, (long)kDh,
          (long)kH * kDh * kDh, (long)kDh * kDh);
    // y[b,h] = q (T x dh) @ att (dh x dh)   -> wxn
    gemmB(wq, watt, wxn, kT, kDh, kDh, (long)kD, 1L, (long)kDh, 1L, (long)kD,
          kB * kH, kH, (long)kT * kD, (long)kDh, (long)kH * kDh * kDh, (long)kDh * kDh,
          (long)kT * kD, (long)kDh);
    gemm(wsemb, sa_ew + l * E2, we, kB, 2 * kD, kTE, sa_eb + (size_t)l * 2 * kD, nullptr, nullptr, 0);
    styl(wxn, we, sa_onw + (size_t)l * kD, sa_onb + (size_t)l * kD, wq);
    gemm(wq, sa_ow + l * DD, wh, kBT, kD, kD, sa_ob + (size_t)l * kD, wh, nullptr, 0);

    // ================= linear temporal cross-attention =================
    ln(wh, wxn, ca_nw + (size_t)l * kD, ca_nb + (size_t)l * kD, kBT, kD);
    ln(xf_out, wxfn, ca_tnw + (size_t)l * kLT, ca_tnb + (size_t)l * kLT, kB * kN, kLT);
    gemm(wxn, ca_qw + l * DD, wq, kBT, kD, kD, ca_qb + (size_t)l * kD, nullptr, nullptr, 0);
    k_softmax_head<<<(int)(((long)kBT * kH + 7) / 8), 256, 0, s>>>(wq, (long)kBT * kH);
    gemm(wxfn, ca_kw + l * LD, wk, kB * kN, kD, kLT, ca_kb + (size_t)l * kD, nullptr, nullptr, 0);
    k_softmax_seq<<<((kB * kD) + 255) / 256, 256, 0, s>>>(wk, nullptr, kB, kN, kD);
    gemm(wxfn, ca_vw + l * LD, wv, kB * kN, kD, kLT, ca_vb + (size_t)l * kD, nullptr, nullptr, 0);
    gemmB(wk, wv, watt, kDh, kDh, kN, 1L, (long)kD, (long)kD, 1L, (long)kDh,
          kB * kH, kH, (long)kN * kD, (long)kDh, (long)kN * kD, (long)kDh,
          (long)kH * kDh * kDh, (long)kDh * kDh);
    gemmB(wq, watt, wxn, kT, kDh, kDh, (long)kD, 1L, (long)kDh, 1L, (long)kD,
          kB * kH, kH, (long)kT * kD, (long)kDh, (long)kH * kDh * kDh, (long)kDh * kDh,
          (long)kT * kD, (long)kDh);
    gemm(wsemb, ca_ew + l * E2, we, kB, 2 * kD, kTE, ca_eb + (size_t)l * 2 * kD, nullptr, nullptr, 0);
    styl(wxn, we, ca_onw + (size_t)l * kD, ca_onb + (size_t)l * kD, wq);
    gemm(wq, ca_ow + l * DD, wh, kBT, kD, kD, ca_ob + (size_t)l * kD, wh, nullptr, 0);

    // ================= FFN =================
    gemm(wh, f1w + (size_t)l * kD * kFF, wff, kBT, kFF, kD, f1b + (size_t)l * kFF,
         nullptr, nullptr, 2);                                           // gelu
    gemm(wff, f2w + (size_t)l * kFF * kD, wxn, kBT, kD, kFF, f2b + (size_t)l * kD,
         nullptr, nullptr, 0);
    gemm(wsemb, fn_ew + l * E2, we, kB, 2 * kD, kTE, fn_eb + (size_t)l * 2 * kD, nullptr, nullptr, 0);
    styl(wxn, we, fn_onw + (size_t)l * kD, fn_onb + (size_t)l * kD, wq);
    gemm(wq, fn_ow + l * DD, wh, kBT, kD, kD, fn_ob + (size_t)l * kD, wh, nullptr, 0);
  }

  // ---- output projection ----
  gemm(wh, ow, out, kBT, kIF, kD, ob, nullptr, nullptr, 0);
}
